// RGN_23862838297329
// MI455X (gfx1250) — matrix operand order repaired and checked by reading, not compiled, not run on hardware
//
#include <hip/hip_runtime.h>

// ---------------------------------------------------------------------------
// RGN (bi-LSTM x2 -> softmax head -> dihedral angles -> NeRF coords) for
// MI455X / gfx1250.  bf16 WMMA (f32 accum) for all GEMMs; persistent
// recurrent kernel with Whh resident in LDS (CDNA5 320KB/WGP), per-step h_t
// broadcast staged via global_load_async_to_lds_b128 (ASYNCcnt path), and a
// grid-wide sense barrier per timestep.  wave32 everywhere.
// ---------------------------------------------------------------------------

typedef __attribute__((ext_vector_type(16))) __bf16 v16bf;
typedef __attribute__((ext_vector_type(2)))  __bf16 bf16x2;
typedef __attribute__((ext_vector_type(8)))  float  v8f;

#define TT 512
#define BB 32
#define HH 800
#define G4 3200          // 4*H
#define H2 1600          // 2*H
#define MM (TT * BB)     // 16384 rows in the projection GEMMs

// LDS layout for the persistent LSTM kernel (bytes)
#define LDS_W     (4 * HH * 16 * 2)        // 102,400: Whh slice, bf16, paired-k layout
#define LDS_G     (4 * BB * 16 * 4)        //   8,192: gate exchange tiles, f32
#define LDS_C     (BB * 16 * 4)            //   2,048: cell state, f32
#define LDS_HST   (BB * HH * 2)            //  51,200: h_t staging, bf16
#define LDS_TOTAL (LDS_W + LDS_G + LDS_C + LDS_HST)   // 163,840 B < 320KB/WGP

__device__ __forceinline__ float sigf(float x) { return 1.0f / (1.0f + __expf(-x)); }

// ---------------------------------------------------------------------------
// Grid-wide sense barrier (counter + generation), used once per LSTM step.
// ---------------------------------------------------------------------------
__device__ __forceinline__ void grid_sync(int* sync) {
    __syncthreads();
    if (threadIdx.x == 0) {
        __threadfence();
        volatile int* gen = sync + 1;
        int g = *gen;
        if (atomicAdd(sync, 1) == (int)gridDim.x - 1) {
            sync[0] = 0;
            __threadfence();
            atomicAdd((int*)(sync + 1), 1);
        } else {
            while (*gen == g) { __builtin_amdgcn_s_sleep(8); }
        }
    }
    __syncthreads();
}

// ---------------------------------------------------------------------------
// Projection GEMM:  xs[M][3200] = A[M][K] * W[3200][K]^T + bias  (bf16 out)
// One wave per 16x16 output tile; v_wmma_f32_16x16x32_bf16 over K.
// ABF=1: A is bf16 (layer-1 input = h1);  ABF=0: A is f32 (layer-0 input x).
// ---------------------------------------------------------------------------
template <int ABF>
__global__ __launch_bounds__(32, 8) void proj_gemm_kernel(
    const void* __restrict__ Aptr, int K,
    const float* __restrict__ W, const float* __restrict__ bias,
    __bf16* __restrict__ xs)
{
    const int tn = blockIdx.x * 16;   // N tile (0..3199)
    const int tm = blockIdx.y * 16;   // M tile (0..16383)
    const int lane = threadIdx.x & 31;
    const int hf = lane >> 4;
    const int lr = lane & 15;

    v8f acc = {};
    const int KP = (K + 31) & ~31;
    const float* wrow = W + (size_t)(tn + lr) * K;   // B fragment: N = lr

    for (int kk = 0; kk < KP; kk += 32) {
        v16bf bfr, afr;
#pragma unroll
        for (int v = 0; v < 8; ++v) {
            int kb = kk + 2 * v + 16 * hf;
            float w0 = (kb     < K) ? wrow[kb]     : 0.0f;
            float w1 = (kb + 1 < K) ? wrow[kb + 1] : 0.0f;
            bfr[2 * v]     = (__bf16)w0;
            bfr[2 * v + 1] = (__bf16)w1;
        }
#pragma unroll
        for (int v = 0; v < 8; ++v) {
            int k0 = kk + 2 * v + 16 * hf;
            float a0, a1;
            if (ABF) {
                const __bf16* ar = (const __bf16*)Aptr + (size_t)(tm + lr) * K;
                a0 = (k0     < K) ? (float)ar[k0]     : 0.0f;
                a1 = (k0 + 1 < K) ? (float)ar[k0 + 1] : 0.0f;
            } else {
                const float* ar = (const float*)Aptr + (size_t)(tm + lr) * K;
                a0 = (k0     < K) ? ar[k0]     : 0.0f;
                a1 = (k0 + 1 < K) ? ar[k0 + 1] : 0.0f;
            }
            afr[2 * v]     = (__bf16)a0;
            afr[2 * v + 1] = (__bf16)a1;
        }
        acc = __builtin_amdgcn_wmma_f32_16x16x32_bf16(
            false, afr, false, bfr, (short)0, acc, false, false);
    }

    const float bv = bias[tn + lr];
#pragma unroll
    for (int r = 0; r < 8; ++r) {
        int row = tm + r + 8 * hf;
        xs[(size_t)row * G4 + tn + lr] = (__bf16)(acc[r] + bv);
    }
}

// ---------------------------------------------------------------------------
// Persistent bi-LSTM layer.  grid = 100 blocks x 128 threads (4 waves).
// blocks 0..49: forward direction; 50..99: backward.  Each block owns 16
// h-columns; Whh slice (4 gates x 16 cols x 800) lives in LDS as bf16 for all
// 512 steps; cell state c lives in LDS.  Per step: h_t (32x800 bf16) is
// staged into LDS once via async global->LDS copies, wave g computes gate g
// for both 16-row batch tiles (2 x 25 WMMAs from LDS), gates exchanged
// through LDS, elementwise update, h written to global (bf16), grid barrier.
// ---------------------------------------------------------------------------
__global__ __launch_bounds__(128, 1) void lstm_layer_kernel(
    const __bf16* __restrict__ xs_f, const __bf16* __restrict__ xs_b,
    const float* __restrict__ Whh_f, const float* __restrict__ Whh_b,
    __bf16* __restrict__ h_out,      // [T][B][1600]
    __bf16* __restrict__ hstate,     // [2][B][800]
    int* __restrict__ sync)
{
    extern __shared__ char smem_raw[];
    __bf16* Wl  = (__bf16*)smem_raw;                            // Whh slice
    float*  Gt  = (float*)(smem_raw + LDS_W);                   // [4][32][16]
    float*  Cst = (float*)(smem_raw + LDS_W + LDS_G);           // [32][16]
    __bf16* Hst = (__bf16*)(smem_raw + LDS_W + LDS_G + LDS_C);  // [32][800] h_t stage

    const int dir    = blockIdx.x / 50;
    const int n0     = (blockIdx.x % 50) * 16;
    const int wave   = threadIdx.x >> 5;       // == gate index
    const int lane   = threadIdx.x & 31;
    const int hf     = lane >> 4;
    const int lr     = lane & 15;

    const float*  Whh = dir ? Whh_b : Whh_f;
    const __bf16* xs  = dir ? xs_b : xs_f;
    __bf16* hs = hstate + (size_t)dir * BB * HH;

    // Stage Whh slice into LDS: layout [g][k>>1][n][k&1] so a (k,k+1) pair for
    // one column is one 32-bit ds load.
    for (int idx = threadIdx.x; idx < 4 * HH * 16; idx += blockDim.x) {
        int n = idx & 15;
        int k = (idx >> 4) % HH;
        int g = idx / (HH * 16);
        float w = Whh[(size_t)(g * HH + n0 + n) * HH + k];
        Wl[g * (HH * 16) + (k >> 1) * 32 + n * 2 + (k & 1)] = (__bf16)w;
    }
    for (int idx = threadIdx.x; idx < BB * 16; idx += blockDim.x) Cst[idx] = 0.0f;
    __syncthreads();

    const int g = wave;
    const __bf16* wb = Wl + g * (HH * 16);
    const unsigned lds_hst = (unsigned)(uintptr_t)Hst;   // wave-relative LDS offset

    for (int step = 0; step < TT; ++step) {
        const int t = dir ? (TT - 1 - step) : step;

        // ---- stage h_{t-1} (32x800 bf16 = 51,200 B) into LDS via the CDNA5
        //      async global->LDS path; tracked by ASYNCcnt.
        {
            const unsigned toff = (unsigned)threadIdx.x * 16u;
#pragma unroll
            for (int i = 0; i < 25; ++i) {
                unsigned off  = toff + (unsigned)i * 2048u;   // 128 thr * 16 B
                unsigned ldst = lds_hst + off;
                asm volatile("global_load_async_to_lds_b128 %0, %1, %2"
                             :: "v"(ldst), "v"(off), "s"(hs)
                             : "memory");
            }
            asm volatile("s_wait_asynccnt 0" ::: "memory");
        }
        __syncthreads();   // LDS h stage visible to all waves

        // prefetch next step's xs tile (global_prefetch_b8 path)
        {
            int tnx = dir ? (t > 0 ? t - 1 : 0) : (t < TT - 1 ? t + 1 : t);
            __builtin_prefetch(xs + ((size_t)tnx * BB + lane) * G4 + g * HH + n0, 0, 1);
        }

        v8f acc0 = {}, acc1 = {};
        for (int kk = 0; kk < HH; kk += 32) {
            v16bf bfr, a0, a1;
#pragma unroll
            for (int v = 0; v < 8; ++v) {
                int kb = kk + 2 * v + 16 * hf;
                bf16x2 p = *(const bf16x2*)(wb + (kb >> 1) * 32 + lr * 2);
                bfr[2 * v] = p[0]; bfr[2 * v + 1] = p[1];
            }
#pragma unroll
            for (int v = 0; v < 8; ++v) {
                int k0 = kk + 2 * v + 16 * hf;
                bf16x2 p0 = *(const bf16x2*)(Hst + lr * HH + k0);
                bf16x2 p1 = *(const bf16x2*)(Hst + (lr + 16) * HH + k0);
                a0[2 * v] = p0[0]; a0[2 * v + 1] = p0[1];
                a1[2 * v] = p1[0]; a1[2 * v + 1] = p1[1];
            }
            acc0 = __builtin_amdgcn_wmma_f32_16x16x32_bf16(
                false, a0, false, bfr, (short)0, acc0, false, false);
            acc1 = __builtin_amdgcn_wmma_f32_16x16x32_bf16(
                false, a1, false, bfr, (short)0, acc1, false, false);
        }

        // add xs pre-activations, stash gate tiles in LDS for exchange
        const int col = g * HH + n0 + lr;
#pragma unroll
        for (int r = 0; r < 8; ++r) {
            int b0 = r + 8 * hf;
            float x0 = (float)xs[((size_t)t * BB + b0)      * G4 + col];
            float x1 = (float)xs[((size_t)t * BB + b0 + 16) * G4 + col];
            Gt[(g * BB + b0)      * 16 + lr] = acc0[r] + x0;
            Gt[(g * BB + b0 + 16) * 16 + lr] = acc1[r] + x1;
        }
        __syncthreads();

        // elementwise LSTM cell update for the 32x16 slice this block owns
        for (int idx = threadIdx.x; idx < BB * 16; idx += blockDim.x) {
            int b = idx >> 4, n = idx & 15;
            float gi = Gt[(0 * BB + b) * 16 + n];
            float gf = Gt[(1 * BB + b) * 16 + n];
            float gg = Gt[(2 * BB + b) * 16 + n];
            float go = Gt[(3 * BB + b) * 16 + n];
            float c  = Cst[idx];
            c = sigf(gf) * c + sigf(gi) * tanhf(gg);
            float h = sigf(go) * tanhf(c);
            Cst[idx] = c;
            __bf16 hb = (__bf16)h;
            hs[(size_t)b * HH + n0 + n] = hb;                               // next-step state
            h_out[((size_t)t * BB + b) * H2 + dir * HH + n0 + n] = hb;      // layer output
        }
        grid_sync(sync);   // publish h_t to all column-blocks of this direction
    }
}

// ---------------------------------------------------------------------------
// Head: logits -> softmax -> dihedral angles.  One wave per (t,b).
// ---------------------------------------------------------------------------
__global__ __launch_bounds__(256, 2) void head_kernel(
    const __bf16* __restrict__ h2, const float* __restrict__ Wl,
    const float* __restrict__ bl, const float* __restrict__ alphabet,
    float* __restrict__ angles)
{
    const int wave = threadIdx.x >> 5;
    const int lane = threadIdx.x & 31;
    const int idx  = blockIdx.x * 8 + wave;         // (t*B + b), 0..16383
    const __bf16* hr = h2 + (size_t)idx * H2;

    float logit[20];
#pragma unroll 4
    for (int a = 0; a < 20; ++a) {
        const float* wr = Wl + a * H2;
        float s = 0.0f;
        for (int k = lane; k < H2; k += 32) s += (float)hr[k] * wr[k];
#pragma unroll
        for (int off = 16; off > 0; off >>= 1) s += __shfl_xor(s, off, 32);
        logit[a] = s + bl[a];
    }
    float mx = logit[0];
#pragma unroll
    for (int a = 1; a < 20; ++a) mx = fmaxf(mx, logit[a]);
    float den = 0.0f;
#pragma unroll
    for (int a = 0; a < 20; ++a) { logit[a] = __expf(logit[a] - mx); den += logit[a]; }
    float inv = 1.0f / den;

    float ys[3] = {0.f, 0.f, 0.f}, xc[3] = {0.f, 0.f, 0.f};
#pragma unroll 4
    for (int a = 0; a < 20; ++a) {
        float p = logit[a] * inv;
#pragma unroll
        for (int j = 0; j < 3; ++j) {
            float al = alphabet[a * 3 + j];
            ys[j] += p * __sinf(al);
            xc[j] += p * __cosf(al);
        }
    }
    if (lane == 0) {
#pragma unroll
        for (int j = 0; j < 3; ++j)
            angles[(size_t)idx * 3 + j] = atan2f(ys[j], xc[j]);
    }
}

// ---------------------------------------------------------------------------
// NeRF coordinate scan: 1536 serial steps, lane = batch element.
// ---------------------------------------------------------------------------
__global__ void coord_kernel(const float* __restrict__ angles, float* __restrict__ out)
{
    const int b = threadIdx.x;                     // 0..31
    const float lens[3] = {1.458f, 1.525f, 1.33f};
    const float angs[3] = {2.124f, 1.941f, 2.028f};
    float ax = 0.f, ay = 0.f, az = 0.f;
    float bx = 1.458f, by = 0.f, bz = 0.f;
    float cx = 2.f, cy = 1.f, cz = 0.f;

    for (int s = 0; s < 3 * TT; ++s) {
        int t = s / 3, j = s - 3 * t;
        float phi = angles[((size_t)t * BB + b) * 3 + j];
        float r = lens[j], th = angs[j];

        float bcx = cx - bx, bcy = cy - by, bcz = cz - bz;
        float iv = rsqrtf(bcx * bcx + bcy * bcy + bcz * bcz);
        bcx *= iv; bcy *= iv; bcz *= iv;
        float abx = bx - ax, aby = by - ay, abz = bz - az;
        float nx = aby * bcz - abz * bcy;
        float ny = abz * bcx - abx * bcz;
        float nz = abx * bcy - aby * bcx;
        iv = rsqrtf(nx * nx + ny * ny + nz * nz);
        nx *= iv; ny *= iv; nz *= iv;
        float mx = ny * bcz - nz * bcy;
        float my = nz * bcx - nx * bcz;
        float mz = nx * bcy - ny * bcx;
        float cp = __cosf(phi), sp = __sinf(phi);
        float ct = __cosf(th),  st = __sinf(th);
        float dx = cx - r * ct * bcx + r * st * (cp * mx + sp * nx);
        float dy = cy - r * ct * bcy + r * st * (cp * my + sp * ny);
        float dz = cz - r * ct * bcz + r * st * (cp * mz + sp * nz);
        out[((size_t)s * BB + b) * 3 + 0] = dx;
        out[((size_t)s * BB + b) * 3 + 1] = dy;
        out[((size_t)s * BB + b) * 3 + 2] = dz;
        ax = bx; ay = by; az = bz;
        bx = cx; by = cy; bz = cz;
        cx = dx; cy = dy; cz = dz;
    }
}

// ---------------------------------------------------------------------------
// Host orchestration
// ---------------------------------------------------------------------------
extern "C" void kernel_launch(void* const* d_in, const int* in_sizes, int n_in,
                              void* d_out, int out_size, void* d_ws, size_t ws_size,
                              hipStream_t stream) {
    (void)in_sizes; (void)n_in; (void)out_size; (void)ws_size;

    const float* x       = (const float*)d_in[0];
    const float* Wih_f0  = (const float*)d_in[1];
    const float* Whh_f0  = (const float*)d_in[2];
    const float* b_f0    = (const float*)d_in[3];
    const float* Wih_b0  = (const float*)d_in[4];
    const float* Whh_b0  = (const float*)d_in[5];
    const float* b_b0    = (const float*)d_in[6];
    const float* Wih_f1  = (const float*)d_in[7];
    const float* Whh_f1  = (const float*)d_in[8];
    const float* b_f1    = (const float*)d_in[9];
    const float* Wih_b1  = (const float*)d_in[10];
    const float* Whh_b1  = (const float*)d_in[11];
    const float* b_b1    = (const float*)d_in[12];
    const float* Wlin    = (const float*)d_in[13];
    const float* bl      = (const float*)d_in[14];
    const float* alpha   = (const float*)d_in[15];
    float* out = (float*)d_out;

    // workspace layout
    const size_t XS_E  = (size_t)TT * BB * G4;   // 52,428,800 bf16
    const size_t H_E   = (size_t)TT * BB * H2;   // 26,214,400 bf16
    char* ws = (char*)d_ws;
    int*     sync   = (int*)ws;                                   // 256 B
    __bf16*  xs_f   = (__bf16*)(ws + 256);
    __bf16*  xs_b   = xs_f + XS_E;
    __bf16*  h1     = xs_b + XS_E;
    __bf16*  h2     = h1 + H_E;
    __bf16*  hstate = h2 + H_E;                                   // 2*32*800 (16B-aligned)
    float*   angles = (float*)(hstate + 2 * BB * HH);             // T*B*3 f32

    hipFuncSetAttribute((const void*)lstm_layer_kernel,
                        hipFuncAttributeMaxDynamicSharedMemorySize, LDS_TOTAL);

    dim3 gproj(G4 / 16, MM / 16);   // 200 x 1024 tiles

    // ---- Layer 0 ----
    hipMemsetAsync(sync, 0, 256, stream);
    hipMemsetAsync(hstate, 0, 2 * BB * HH * sizeof(__bf16), stream);
    proj_gemm_kernel<0><<<gproj, 32, 0, stream>>>((const void*)x, 41, Wih_f0, b_f0, xs_f);
    proj_gemm_kernel<0><<<gproj, 32, 0, stream>>>((const void*)x, 41, Wih_b0, b_b0, xs_b);
    lstm_layer_kernel<<<100, 128, LDS_TOTAL, stream>>>(xs_f, xs_b, Whh_f0, Whh_b0,
                                                       h1, hstate, sync);

    // ---- Layer 1 ----
    hipMemsetAsync(sync, 0, 256, stream);
    hipMemsetAsync(hstate, 0, 2 * BB * HH * sizeof(__bf16), stream);
    proj_gemm_kernel<1><<<gproj, 32, 0, stream>>>((const void*)h1, H2, Wih_f1, b_f1, xs_f);
    proj_gemm_kernel<1><<<gproj, 32, 0, stream>>>((const void*)h1, H2, Wih_b1, b_b1, xs_b);
    lstm_layer_kernel<<<100, 128, LDS_TOTAL, stream>>>(xs_f, xs_b, Whh_f1, Whh_b1,
                                                       h2, hstate, sync);

    // ---- Head + geometry ----
    head_kernel<<<MM / 8, 256, 0, stream>>>(h2, Wlin, bl, alpha, angles);
    coord_kernel<<<1, 32, 0, stream>>>(angles, out);
}
